// LSTMCell_24455543783999
// MI455X (gfx1250) — matrix operand order repaired and checked by reading, not compiled, not run on hardware
//
#include <hip/hip_runtime.h>
#include <hip/hip_bf16.h>

// LSTM cell on MI455X (gfx1250), two paths:
//  FAST (needs 96MB workspace): one-shot fp32->bf16 convert of weights (pre-transposed
//    [n][k]) and x/h into d_ws, then a bf16 WMMA GEMM whose tiles are moved
//    global->LDS with GLOBAL_LOAD_ASYNC_TO_LDS_B128 (ASYNCcnt) into a double buffer.
//  FALLBACK: fused fp32->bf16-in-register kernel (previous round's design).

#define BATCH  4096
#define INPUT  2048
#define HIDDEN 2048

#define BM 128
#define BN 64
#define KT 32
#define LDS_PITCH 40   // 32 bf16 + 8 pad -> 80B row stride, conflict-free 16-lane access
#define KSTEPS (INPUT / KT)          // 64 per half
#define NSTEP  (2 * KSTEPS)          // 128 total (x-half then h-half)

typedef __attribute__((ext_vector_type(16))) __bf16  v16bf;
typedef __attribute__((ext_vector_type(8)))  float   v8f;
typedef __attribute__((ext_vector_type(4)))  unsigned u32x4;
typedef __attribute__((ext_vector_type(4)))  float   f32x4;

union FragCast { u32x4 q[2]; v16bf v; };

__device__ __forceinline__ unsigned pack_bf16(float a, float b) {
  unsigned ua = __builtin_bit_cast(unsigned, a);
  unsigned ub = __builtin_bit_cast(unsigned, b);
  // round-to-nearest-even f32 -> bf16
  ua = (ua + 0x7FFFu + ((ua >> 16) & 1u)) >> 16;
  ub = (ub + 0x7FFFu + ((ub >> 16) & 1u)) >> 16;
  return (ua & 0xFFFFu) | (ub << 16);
}

// Overflow-safe fast activations on v_exp_f32 / v_rcp_f32.
__device__ __forceinline__ float fast_sigmoid(float v) {
  return __builtin_amdgcn_rcpf(1.0f + __expf(-v));
}
__device__ __forceinline__ float fast_tanh(float v) {
  // tanh(v) = 1 - 2/(exp(2v)+1): exp->inf gives +1, exp->0 gives -1. No NaN.
  return 1.0f - 2.0f * __builtin_amdgcn_rcpf(__expf(2.0f * v) + 1.0f);
}

// ---- CDNA5 async global->LDS copy (16B), tracked by ASYNCcnt ----
__device__ __forceinline__ void async_copy16(const unsigned short* gsrc,
                                             unsigned short* ldst) {
  // Flat address of a __shared__ object carries the LDS byte offset in addr[31:0].
  unsigned loff = (unsigned)(unsigned long long)(uintptr_t)ldst;
  unsigned long long gaddr = (unsigned long long)(uintptr_t)gsrc;
  asm volatile("global_load_async_to_lds_b128 %0, %1, off"
               :: "v"(loff), "v"(gaddr) : "memory");
}
__device__ __forceinline__ void wait_async0() {
  asm volatile("s_wait_asynccnt 0" ::: "memory");
}

// A fragment, 16x32 bf16 (ISA 7.12.2): lane<16: M=lane, K=0..7 in v0..3, K=16..23 in v4..7;
// lane>=16: same M, K=8..15 and K=24..31.  base = row 0 of a 16-row LDS sub-tile [row][k].
__device__ __forceinline__ v16bf load_frag_a(const unsigned short* base, int lane) {
  int row  = lane & 15;
  int koff = (lane >> 4) << 3;              // 0 or 8
  const unsigned short* p = base + row * LDS_PITCH + koff;
  FragCast f;
  f.q[0] = *(const u32x4*)(p);              // K = koff .. koff+7
  f.q[1] = *(const u32x4*)(p + 16);         // K = koff+16 .. koff+23
  return f.v;
}

// B fragment, 32x16 bf16 stored transposed in LDS as [n][k]:
// lanes 0-15 hold N=lane with K=0..7 (v0..3), K=16..23 (v4..7); lanes 16-31 hold K=8..15, K=24..31.
__device__ __forceinline__ v16bf load_frag_b(const unsigned short* base, int lane) {
  int ncol = lane & 15;
  int koff = (lane >> 4) << 3;              // 0 or 8
  const unsigned short* p = base + ncol * LDS_PITCH + koff;
  FragCast f;
  f.q[0] = *(const u32x4*)(p);
  f.q[1] = *(const u32x4*)(p + 16);
  return f.v;
}

// ===================== one-shot converters (fast path) =====================

// Weights: 8 matrices (w_ih[0..3], w_hh[0..3]) of [K=2048][N=2048] f32
//   -> bf16 pre-transposed [N=2048][K=2048] via LDS-tiled transpose.
__global__ __launch_bounds__(256)
void convert_wt_kernel(const float* __restrict__ w_ih,
                       const float* __restrict__ w_hh,
                       unsigned short* __restrict__ wt) {
  __shared__ float tile[64][65];            // 65 pitch: conflict-free transposed reads
  const int z = blockIdx.z;                 // 0..7
  const float* src = (z < 4) ? (w_ih + (size_t)z * INPUT * HIDDEN)
                             : (w_hh + (size_t)(z - 4) * INPUT * HIDDEN);
  unsigned short* dst = wt + (size_t)z * INPUT * HIDDEN;
  const int k0 = blockIdx.x * 64;
  const int n0 = blockIdx.y * 64;
  const int t  = threadIdx.x;

  #pragma unroll
  for (int i = 0; i < 4; ++i) {             // coalesced 64x64 f32 tile read
    int cidx = t + i * 256;                 // 0..1023
    int r  = cidx >> 4;
    int c4 = (cidx & 15) * 4;
    f32x4 v = *(const f32x4*)(src + (size_t)(k0 + r) * HIDDEN + n0 + c4);
    tile[r][c4 + 0] = v.x; tile[r][c4 + 1] = v.y;
    tile[r][c4 + 2] = v.z; tile[r][c4 + 3] = v.w;
  }
  __syncthreads();
  #pragma unroll
  for (int i = 0; i < 2; ++i) {             // coalesced transposed bf16 write
    int cidx = t + i * 256;                 // 0..511
    int n  = cidx >> 3;
    int ks = (cidx & 7) * 8;
    u32x4 p = { pack_bf16(tile[ks + 0][n], tile[ks + 1][n]),
                pack_bf16(tile[ks + 2][n], tile[ks + 3][n]),
                pack_bf16(tile[ks + 4][n], tile[ks + 5][n]),
                pack_bf16(tile[ks + 6][n], tile[ks + 7][n]) };
    *(u32x4*)(dst + (size_t)(n0 + n) * INPUT + k0 + ks) = p;
  }
}

// Activations x,h: elementwise fp32 -> bf16.
__global__ __launch_bounds__(256)
void convert_a_kernel(const float* __restrict__ x, const float* __restrict__ h,
                      unsigned short* __restrict__ ab) {
  const float* src = blockIdx.y ? h : x;
  unsigned short* dst = ab + (size_t)blockIdx.y * BATCH * INPUT;
  size_t i = ((size_t)blockIdx.x * 256 + threadIdx.x) * 8;
  f32x4 a = *(const f32x4*)(src + i);
  f32x4 b = *(const f32x4*)(src + i + 4);
  u32x4 p = { pack_bf16(a.x, a.y), pack_bf16(a.z, a.w),
              pack_bf16(b.x, b.y), pack_bf16(b.z, b.w) };
  *(u32x4*)(dst + i) = p;
}

// ===================== fast path: bf16 WMMA GEMM with async-LDS =====================

__global__ __launch_bounds__(256, 1)
void lstm_wmma_bf16(const unsigned short* __restrict__ ab,   // [2][BATCH][INPUT] bf16 (x,h)
                    const unsigned short* __restrict__ wt,   // [8][N=2048][K=2048] bf16
                    const float* __restrict__ c,
                    const float* __restrict__ b_ih,
                    const float* __restrict__ b_hh,
                    float* __restrict__ out) {
  __shared__ __align__(16) unsigned short Abuf[2][BM * LDS_PITCH];        // 2 x 10 KB
  __shared__ __align__(16) unsigned short Bbuf[2][4][BN * LDS_PITCH];     // 2 x 20 KB

  const int tid  = threadIdx.x;
  const int lane = tid & 31;
  const int wave = tid >> 5;
  const int wave_m = wave & 3;
  const int wave_n = wave >> 2;

  const int m0 = blockIdx.x * BM;
  const int n0 = blockIdx.y * BN;

  v8f acc[4][2][2];
  #pragma unroll
  for (int g = 0; g < 4; ++g)
    #pragma unroll
    for (int mi = 0; mi < 2; ++mi)
      #pragma unroll
      for (int ni = 0; ni < 2; ++ni)
        acc[g][mi][ni] = (v8f){0.f, 0.f, 0.f, 0.f, 0.f, 0.f, 0.f, 0.f};

  // copy assignments (16B chunks): A tile 128x32 bf16 = 512 chunks -> 2/thread
  const int a_row = tid >> 1;
  const int a_sg  = (tid & 1) * 2;          // chunk pair 0-1 or 2-3
  // B tile per gate 64x32 bf16 = 256 chunks -> 1/thread/gate
  const int b_n  = tid >> 2;
  const int b_sg = tid & 3;

  auto issue_copies = [&](int stg, int step) {
    const unsigned short* asrc = ab + ((step >= KSTEPS) ? (size_t)BATCH * INPUT : 0);
    const unsigned short* wsrc = wt + ((step >= KSTEPS) ? (size_t)4 * INPUT * HIDDEN : 0);
    const int k0 = (step & (KSTEPS - 1)) * KT;
    // A: row-major [m][k]
    {
      const unsigned short* g0 = asrc + (size_t)(m0 + a_row) * INPUT + k0 + a_sg * 8;
      unsigned short* l0 = &Abuf[stg][a_row * LDS_PITCH + a_sg * 8];
      async_copy16(g0,     l0);
      async_copy16(g0 + 8, l0 + 8);
    }
    // B: already transposed in global -> contiguous rows [n][k]
    #pragma unroll
    for (int g = 0; g < 4; ++g) {
      const unsigned short* gs = wsrc + (size_t)g * INPUT * HIDDEN
                               + (size_t)(n0 + b_n) * INPUT + k0 + b_sg * 8;
      async_copy16(gs, &Bbuf[stg][g][b_n * LDS_PITCH + b_sg * 8]);
    }
  };

  issue_copies(0, 0);

  for (int step = 0; step < NSTEP; ++step) {
    const int cur = step & 1;

    wait_async0();       // this wave's copies into stage[cur] complete
    __syncthreads();     // all waves' copies done; all waves done reading stage[cur^1]

    if (step + 1 < NSTEP) issue_copies(cur ^ 1, step + 1);   // overlaps WMMAs below

    v16bf afrag[2];
    afrag[0] = load_frag_a(&Abuf[cur][(wave_m * 32 + 0)  * LDS_PITCH], lane);
    afrag[1] = load_frag_a(&Abuf[cur][(wave_m * 32 + 16) * LDS_PITCH], lane);

    #pragma unroll
    for (int g = 0; g < 4; ++g) {
      v16bf bf0 = load_frag_b(&Bbuf[cur][g][(wave_n * 32 + 0)  * LDS_PITCH], lane);
      v16bf bf1 = load_frag_b(&Bbuf[cur][g][(wave_n * 32 + 16) * LDS_PITCH], lane);
      acc[g][0][0] = __builtin_amdgcn_wmma_f32_16x16x32_bf16(
          false, afrag[0], false, bf0, (short)0, acc[g][0][0], false, false);
      acc[g][0][1] = __builtin_amdgcn_wmma_f32_16x16x32_bf16(
          false, afrag[0], false, bf1, (short)0, acc[g][0][1], false, false);
      acc[g][1][0] = __builtin_amdgcn_wmma_f32_16x16x32_bf16(
          false, afrag[1], false, bf0, (short)0, acc[g][1][0], false, false);
      acc[g][1][1] = __builtin_amdgcn_wmma_f32_16x16x32_bf16(
          false, afrag[1], false, bf1, (short)0, acc[g][1][1], false, false);
    }
  }

  // ---- epilogue: bias + activations + state update ----
  float* __restrict__ out_h = out;
  float* __restrict__ out_c = out + (size_t)BATCH * HIDDEN;

  #pragma unroll
  for (int ni = 0; ni < 2; ++ni) {
    const int ncol = n0 + wave_n * 32 + ni * 16 + (lane & 15);
    const float bi = b_ih[0 * HIDDEN + ncol] + b_hh[0 * HIDDEN + ncol];
    const float bf = b_ih[1 * HIDDEN + ncol] + b_hh[1 * HIDDEN + ncol];
    const float bg = b_ih[2 * HIDDEN + ncol] + b_hh[2 * HIDDEN + ncol];
    const float bo = b_ih[3 * HIDDEN + ncol] + b_hh[3 * HIDDEN + ncol];
    #pragma unroll
    for (int mi = 0; mi < 2; ++mi) {
      const int rowbase = m0 + wave_m * 32 + mi * 16 + ((lane >> 4) * 8);
      #pragma unroll
      for (int r = 0; r < 8; ++r) {
        const int row = rowbase + r;
        const size_t idx = (size_t)row * HIDDEN + ncol;
        float gi = fast_sigmoid(acc[0][mi][ni][r] + bi);
        float gf = fast_sigmoid(acc[1][mi][ni][r] + bf);
        float gg = fast_tanh   (acc[2][mi][ni][r] + bg);
        float go = fast_sigmoid(acc[3][mi][ni][r] + bo);
        float cold = c[idx];
        float cn = gf * cold + gi * gg;
        float hn = go * fast_tanh(cn);
        out_h[idx] = hn;
        out_c[idx] = cn;
      }
    }
  }
}

// ===================== fallback: fused fp32-input kernel =====================

__global__ __launch_bounds__(256, 1)
void lstm_cell_wmma_fused(const float* __restrict__ x,
                          const float* __restrict__ h,
                          const float* __restrict__ c,
                          const float* __restrict__ w_ih,
                          const float* __restrict__ w_hh,
                          const float* __restrict__ b_ih,
                          const float* __restrict__ b_hh,
                          float* __restrict__ out) {
  __shared__ __align__(16) unsigned short Abuf[BM * LDS_PITCH];
  __shared__ __align__(16) unsigned short Bbuf[4][BN * LDS_PITCH];

  const int tid  = threadIdx.x;
  const int lane = tid & 31;
  const int wave = tid >> 5;
  const int wave_m = wave & 3;
  const int wave_n = wave >> 2;

  const int m0 = blockIdx.x * BM;
  const int n0 = blockIdx.y * BN;

  v8f acc[4][2][2];
  #pragma unroll
  for (int g = 0; g < 4; ++g)
    #pragma unroll
    for (int mi = 0; mi < 2; ++mi)
      #pragma unroll
      for (int ni = 0; ni < 2; ++ni)
        acc[g][mi][ni] = (v8f){0.f, 0.f, 0.f, 0.f, 0.f, 0.f, 0.f, 0.f};

  const int a_row  = tid >> 1;
  const int a_cseg = (tid & 1) * 16;
  const int b_kb = (tid >> 4) * 2;
  const int b_nb = (tid & 15) * 4;

  f32x4 sa[4];
  f32x4 sb[4][2];

  auto issue_loads = [&](int step) {
    const float* __restrict__ Am = (step < KSTEPS) ? x    : h;
    const float* __restrict__ Wm = (step < KSTEPS) ? w_ih : w_hh;
    const int k0 = (step & (KSTEPS - 1)) * KT;
    const float* gp = Am + (size_t)(m0 + a_row) * INPUT + k0 + a_cseg;
    sa[0] = *(const f32x4*)(gp);
    sa[1] = *(const f32x4*)(gp + 4);
    sa[2] = *(const f32x4*)(gp + 8);
    sa[3] = *(const f32x4*)(gp + 12);
    #pragma unroll
    for (int g = 0; g < 4; ++g) {
      const float* wp = Wm + (size_t)g * INPUT * HIDDEN
                           + (size_t)(k0 + b_kb) * HIDDEN + n0 + b_nb;
      sb[g][0] = *(const f32x4*)(wp);
      sb[g][1] = *(const f32x4*)(wp + HIDDEN);
    }
  };

  issue_loads(0);

  for (int step = 0; step < NSTEP; ++step) {
    __syncthreads();
    {
      u32x4 p0 = { pack_bf16(sa[0].x, sa[0].y), pack_bf16(sa[0].z, sa[0].w),
                   pack_bf16(sa[1].x, sa[1].y), pack_bf16(sa[1].z, sa[1].w) };
      u32x4 p1 = { pack_bf16(sa[2].x, sa[2].y), pack_bf16(sa[2].z, sa[2].w),
                   pack_bf16(sa[3].x, sa[3].y), pack_bf16(sa[3].z, sa[3].w) };
      *(u32x4*)&Abuf[a_row * LDS_PITCH + a_cseg]     = p0;
      *(u32x4*)&Abuf[a_row * LDS_PITCH + a_cseg + 8] = p1;
    }
    #pragma unroll
    for (int g = 0; g < 4; ++g) {
      unsigned short* bl = &Bbuf[g][0];
      *(unsigned*)&bl[(b_nb + 0) * LDS_PITCH + b_kb] = pack_bf16(sb[g][0].x, sb[g][1].x);
      *(unsigned*)&bl[(b_nb + 1) * LDS_PITCH + b_kb] = pack_bf16(sb[g][0].y, sb[g][1].y);
      *(unsigned*)&bl[(b_nb + 2) * LDS_PITCH + b_kb] = pack_bf16(sb[g][0].z, sb[g][1].z);
      *(unsigned*)&bl[(b_nb + 3) * LDS_PITCH + b_kb] = pack_bf16(sb[g][0].w, sb[g][1].w);
    }
    __syncthreads();
    if (step + 1 < NSTEP) issue_loads(step + 1);

    v16bf afrag[2];
    afrag[0] = load_frag_a(&Abuf[(wave_m * 32 + 0)  * LDS_PITCH], lane);
    afrag[1] = load_frag_a(&Abuf[(wave_m * 32 + 16) * LDS_PITCH], lane);

    #pragma unroll
    for (int g = 0; g < 4; ++g) {
      v16bf bf0 = load_frag_b(&Bbuf[g][(wave_n * 32 + 0)  * LDS_PITCH], lane);
      v16bf bf1 = load_frag_b(&Bbuf[g][(wave_n * 32 + 16) * LDS_PITCH], lane);
      acc[g][0][0] = __builtin_amdgcn_wmma_f32_16x16x32_bf16(
          false, afrag[0], false, bf0, (short)0, acc[g][0][0], false, false);
      acc[g][0][1] = __builtin_amdgcn_wmma_f32_16x16x32_bf16(
          false, afrag[0], false, bf1, (short)0, acc[g][0][1], false, false);
      acc[g][1][0] = __builtin_amdgcn_wmma_f32_16x16x32_bf16(
          false, afrag[1], false, bf0, (short)0, acc[g][1][0], false, false);
      acc[g][1][1] = __builtin_amdgcn_wmma_f32_16x16x32_bf16(
          false, afrag[1], false, bf1, (short)0, acc[g][1][1], false, false);
    }
  }

  float* __restrict__ out_h = out;
  float* __restrict__ out_c = out + (size_t)BATCH * HIDDEN;

  #pragma unroll
  for (int ni = 0; ni < 2; ++ni) {
    const int ncol = n0 + wave_n * 32 + ni * 16 + (lane & 15);
    const float bi = b_ih[0 * HIDDEN + ncol] + b_hh[0 * HIDDEN + ncol];
    const float bf = b_ih[1 * HIDDEN + ncol] + b_hh[1 * HIDDEN + ncol];
    const float bg = b_ih[2 * HIDDEN + ncol] + b_hh[2 * HIDDEN + ncol];
    const float bo = b_ih[3 * HIDDEN + ncol] + b_hh[3 * HIDDEN + ncol];
    #pragma unroll
    for (int mi = 0; mi < 2; ++mi) {
      const int rowbase = m0 + wave_m * 32 + mi * 16 + ((lane >> 4) * 8);
      #pragma unroll
      for (int r = 0; r < 8; ++r) {
        const int row = rowbase + r;
        const size_t idx = (size_t)row * HIDDEN + ncol;
        float gi = fast_sigmoid(acc[0][mi][ni][r] + bi);
        float gf = fast_sigmoid(acc[1][mi][ni][r] + bf);
        float gg = fast_tanh   (acc[2][mi][ni][r] + bg);
        float go = fast_sigmoid(acc[3][mi][ni][r] + bo);
        float cold = c[idx];
        float cn = gf * cold + gi * gg;
        float hn = go * fast_tanh(cn);
        out_h[idx] = hn;
        out_c[idx] = cn;
      }
    }
  }
}

extern "C" void kernel_launch(void* const* d_in, const int* in_sizes, int n_in,
                              void* d_out, int out_size, void* d_ws, size_t ws_size,
                              hipStream_t stream) {
  (void)in_sizes; (void)n_in; (void)out_size;
  const float* x    = (const float*)d_in[0];
  const float* h    = (const float*)d_in[1];
  const float* c    = (const float*)d_in[2];
  const float* w_ih = (const float*)d_in[3];
  const float* w_hh = (const float*)d_in[4];
  const float* b_ih = (const float*)d_in[5];
  const float* b_hh = (const float*)d_in[6];
  float* out = (float*)d_out;

  const size_t WT_BYTES = (size_t)8 * INPUT * HIDDEN * 2;   // 64 MB bf16 weights
  const size_t AB_BYTES = (size_t)2 * BATCH * INPUT * 2;    // 32 MB bf16 x,h
  dim3 grid(BATCH / BM, HIDDEN / BN);                       // 32 x 32

  if (ws_size >= WT_BYTES + AB_BYTES) {
    unsigned short* wt = (unsigned short*)d_ws;
    unsigned short* ab = (unsigned short*)((char*)d_ws + WT_BYTES);
    convert_wt_kernel<<<dim3(INPUT / 64, HIDDEN / 64, 8), 256, 0, stream>>>(w_ih, w_hh, wt);
    convert_a_kernel<<<dim3((BATCH * INPUT) / 2048, 2), 256, 0, stream>>>(x, h, ab);
    lstm_wmma_bf16<<<grid, 256, 0, stream>>>(ab, wt, c, b_ih, b_hh, out);
  } else {
    lstm_cell_wmma_fused<<<grid, 256, 0, stream>>>(x, h, c, w_ih, w_hh, b_ih, b_hh, out);
  }
}
